// LightGCN_52596169507041
// MI455X (gfx1250) — matrix operand order repaired and checked by reading, not compiled, not run on hardware
//
#include <hip/hip_runtime.h>
#include <hip/hip_bf16.h>
#include <cstddef>

#define NUM_USERS 200000
#define NUM_ITEMS 100000
#define EMBED_DIM 64
#define N_NODES   (NUM_USERS + NUM_ITEMS)
#define NNZ       6000000
#define BATCH     16384

typedef __attribute__((ext_vector_type(16))) _Float16 v16h;
typedef __attribute__((ext_vector_type(8)))  float    v8f;

union V16H { v16h v; _Float16 h[16]; };
union V8F  { v8f  v; float    f[8];  };

// ---------------------------------------------------------------------------
// Kernel 1: concat user/item embeddings into x (buf0) and acc, float4 wide.
// ---------------------------------------------------------------------------
__global__ void lgcn_init_concat(const float* __restrict__ user_emb,
                                 const float* __restrict__ item_emb,
                                 float* __restrict__ x,
                                 float* __restrict__ acc) {
    const long long n4 = (long long)N_NODES * (EMBED_DIM / 4);       // float4 count
    const long long u4 = (long long)NUM_USERS * (EMBED_DIM / 4);
    long long i = (long long)blockIdx.x * blockDim.x + threadIdx.x;
    if (i >= n4) return;
    float4 v;
    if (i < u4) v = ((const float4*)user_emb)[i];
    else        v = ((const float4*)item_emb)[i - u4];
    ((float4*)x)[i]   = v;
    ((float4*)acc)[i] = v;
}

// ---------------------------------------------------------------------------
// Kernel 2: zero the next-layer buffer, float4 wide.
// ---------------------------------------------------------------------------
__global__ void lgcn_zero(float* __restrict__ p) {
    const long long n4 = (long long)N_NODES * (EMBED_DIM / 4);
    long long i = (long long)blockIdx.x * blockDim.x + threadIdx.x;
    if (i >= n4) return;
    ((float4*)p)[i] = make_float4(0.f, 0.f, 0.f, 0.f);
}

// ---------------------------------------------------------------------------
// Kernel 3: edge scatter SpMM.  16 threads per edge, float4 per thread:
//   x_next[row, :] += val * x[col, :]
// Gather is a coalesced 128-bit load; scatter is 4x global_atomic_add_f32
// (L2-resident for this working set: x + x_next + edges ~ 225 MB vs 192 MB L2).
// ---------------------------------------------------------------------------
__global__ void lgcn_spmm(const int*   __restrict__ rows,
                          const int*   __restrict__ cols,
                          const float* __restrict__ vals,
                          const float* __restrict__ x,
                          float*       __restrict__ xnext) {
    long long t = (long long)blockIdx.x * blockDim.x + threadIdx.x;
    long long e = t >> 4;          // edge index
    int sub = (int)(t & 15);       // which float4 of the 64-dim row
    if (e >= NNZ) return;
    int   c = cols[e];
    int   r = rows[e];
    float v = vals[e];
    float4 m = ((const float4*)(x + (size_t)c * EMBED_DIM))[sub];
    float* dst = xnext + (size_t)r * EMBED_DIM + sub * 4;
    atomicAdd(dst + 0, v * m.x);
    atomicAdd(dst + 1, v * m.y);
    atomicAdd(dst + 2, v * m.z);
    atomicAdd(dst + 3, v * m.w);
}

// ---------------------------------------------------------------------------
// Kernel 4: acc += x, float4 wide.
// ---------------------------------------------------------------------------
__global__ void lgcn_acc_add(float* __restrict__ acc, const float* __restrict__ x) {
    const long long n4 = (long long)N_NODES * (EMBED_DIM / 4);
    long long i = (long long)blockIdx.x * blockDim.x + threadIdx.x;
    if (i >= n4) return;
    float4 a = ((float4*)acc)[i];
    float4 b = ((const float4*)x)[i];
    a.x += b.x; a.y += b.y; a.z += b.z; a.w += b.w;
    ((float4*)acc)[i] = a;
}

// ---------------------------------------------------------------------------
// Kernel 5: WMMA scoring. One wave32 handles 16 (user,item) pairs.
//   A = U-tile (16x32 f16), B = I^T-tile (32x16 f16), two k-chunks of 32.
//   scores = diagonal of D (f32 accumulator). The /4 layer-mean is folded
//   into the f16 conversion; diag(U/4 * (I/4)^T) = mean-dot exactly.
//
// VGPR layouts per CDNA5 ISA:
//  A (16-bit 16x32): lane l (l<16) row M=l,    halves t: K = t<8 ? t   : t+8
//                    lane l (>=16) row M=l-16, halves t: K = t<8 ? t+8 : t+16
//  B (16-bit 32x16): lane l (l<16) col N=l,    halves t: K = t<8 ? t   : t+8
//                    lane l (>=16) col N=l-16, halves t: K = t<8 ? t+8 : t+16
//  C/D (32-bit 16x16): lanes 0-15 col N=l rows 0-7 in v0-7;
//                      lanes 16-31 col N=l-16 rows 8-15 in v0-7.
// ---------------------------------------------------------------------------
__global__ void lgcn_score_wmma(const float* __restrict__ acc,
                                const int*   __restrict__ user_ids,
                                const int*   __restrict__ item_ids,
                                float*       __restrict__ scores) {
    const int lane  = threadIdx.x;          // blockDim.x == 32
    const int lo    = lane & 15;
    const bool hiH  = lane >= 16;
    const int base  = blockIdx.x * 16;      // 16 pairs per wave
    const int pair  = base + lo;

    const float* urow = acc + (size_t)user_ids[pair] * EMBED_DIM;
    const float* irow = acc + (size_t)(NUM_USERS + item_ids[pair]) * EMBED_DIM;
    const float scale = 0.25f;              // 1/(N_LAYERS+1)

    V8F c; c.v = (v8f){0.f, 0.f, 0.f, 0.f, 0.f, 0.f, 0.f, 0.f};

    #pragma unroll
    for (int chunk = 0; chunk < 2; ++chunk) {
        const int koff = chunk * 32;
        V16H a, b;
        #pragma unroll
        for (int t = 0; t < 16; ++t) {
            int ka = hiH ? ((t < 8) ? t + 8 : t + 16)
                         : ((t < 8) ? t     : t + 8);
            int kb = ka;
            a.h[t] = (_Float16)(scale * urow[koff + ka]);
            b.h[t] = (_Float16)(scale * irow[koff + kb]);
        }
        // emits v_wmma_f32_16x16x32_f16
        c.v = __builtin_amdgcn_wmma_f32_16x16x32_f16(
            /*neg_a=*/false, a.v, /*neg_b=*/false, b.v,
            /*c_mod=*/(short)0, c.v, /*reuse_a=*/false, /*reuse_b=*/false);
    }

    // Diagonal D[p][p]:
    //   p<8 : held by lane p    in c.f[p]
    //   p>=8: held by lane p+16 in c.f[p-8]
    // Branch-free gather of c.f[lane & 7] via a cndmask select chain, then a
    // single predicated store (avoids the 32-way exec-mask branch tree).
    const int r = lane & 7;
    float val = c.f[0];
    #pragma unroll
    for (int j = 1; j < 8; ++j) val = (r == j) ? c.f[j] : val;

    const bool writer = (lane < 8) || (lane >= 24);
    const int  outIdx = base + ((lane < 8) ? lane : lane - 16);
    if (writer) scores[outIdx] = val;
}

// ---------------------------------------------------------------------------
// Host launcher. Workspace layout (needs >= 3 * N_NODES*64*4 B ~ 230 MB):
//   [buf0 | buf1 | acc]
// ---------------------------------------------------------------------------
extern "C" void kernel_launch(void* const* d_in, const int* in_sizes, int n_in,
                              void* d_out, int out_size, void* d_ws, size_t ws_size,
                              hipStream_t stream) {
    const float* user_emb = (const float*)d_in[0];
    const float* item_emb = (const float*)d_in[1];
    const int*   adj_rows = (const int*)  d_in[2];
    const int*   adj_cols = (const int*)  d_in[3];
    const float* adj_vals = (const float*)d_in[4];
    const int*   user_ids = (const int*)  d_in[5];
    const int*   item_ids = (const int*)  d_in[6];
    float*       scores   = (float*)d_out;

    const size_t nodeF = (size_t)N_NODES * EMBED_DIM;   // floats per buffer
    float* buf0 = (float*)d_ws;
    float* buf1 = buf0 + nodeF;
    float* acc  = buf1 + nodeF;

    const long long n4 = (long long)N_NODES * (EMBED_DIM / 4);
    const int elemBlocks = (int)((n4 + 255) / 256);
    const long long spmmThreads = (long long)NNZ * 16;
    const int spmmBlocks = (int)((spmmThreads + 255) / 256);

    lgcn_init_concat<<<elemBlocks, 256, 0, stream>>>(user_emb, item_emb, buf0, acc);

    float* bufs[2] = {buf0, buf1};
    for (int layer = 0; layer < 3; ++layer) {
        float* cur = bufs[layer & 1];
        float* nxt = bufs[(layer + 1) & 1];
        lgcn_zero<<<elemBlocks, 256, 0, stream>>>(nxt);
        lgcn_spmm<<<spmmBlocks, 256, 0, stream>>>(adj_rows, adj_cols, adj_vals, cur, nxt);
        lgcn_acc_add<<<elemBlocks, 256, 0, stream>>>(acc, nxt);
    }

    lgcn_score_wmma<<<BATCH / 16, 32, 0, stream>>>(acc, user_ids, item_ids, scores);
}
